// BlockSparseAttentionWrapper_82205674045542
// MI455X (gfx1250) — matrix operand order repaired and checked by reading, not compiled, not run on hardware
//
#include <hip/hip_runtime.h>
#include <hip/hip_bf16.h>
#include <math.h>

typedef __attribute__((ext_vector_type(16))) _Float16 v16h;
typedef __attribute__((ext_vector_type(8)))  _Float16 v8h;
typedef __attribute__((ext_vector_type(8)))  float    v8f;

#define Bc   32
#define Nc   1024
#define Cc   768
#define Hc   12
#define Dc   64      // head dim
#define BSc  32      // block size
#define MROWS (Bc * Nc)      // 32768
#define C3   (3 * Cc)        // 2304

// LDS weight panel: 64 rows x K halves, padded row stride.
// Pad = 8 halves -> stride 776 halves = 388 dwords; 388 % 64 = 4, so the 16
// lanes of a B-fragment (stride 776*n) land on banks 4n: conflict-free b128s.
#define PADK (Cc + 8)            // 776 halves
#define SB_BYTES (64 * PADK * 2) // 99,328 bytes of LDS

__device__ __forceinline__ v8f wmma_f16(v16h a, v16h b, v8f c) {
    return __builtin_amdgcn_wmma_f32_16x16x32_f16(
        /*neg_a=*/false, a, /*neg_b=*/false, b,
        /*c_mod=*/(short)0, c, /*reuse_a=*/false, /*reuse_b=*/false);
}

// A fragment 16x32 f16 from row-major memory (contiguous K order, matches B):
// lanes 0-15: row = lane,    K in {0..15}
// lanes 16-31: row = lane-16, K in {16..31}
__device__ __forceinline__ v16h load_a_frag(const _Float16* __restrict__ base,
                                            int ld, int lane) {
    int r  = lane & 15;
    int kk = (lane & 16) ? 16 : 0;
    const _Float16* p = base + (size_t)r * ld + kk;
    v8h lo = *(const v8h*)p;          // K = kk .. kk+7
    v8h hi = *(const v8h*)(p + 8);    // K = kk+8 .. kk+15
    v16h out;
#pragma unroll
    for (int i = 0; i < 8; ++i) { out[i] = lo[i]; out[i + 8] = hi[i]; }
    return out;
}

// B fragment 32x16 f16 from row-major W (B[k][n] = W[n][k]):
// lanes 0-15: col = lane, K=0..15 ; lanes 16-31: col = lane-16, K=16..31
__device__ __forceinline__ v16h load_b_frag_g(const _Float16* __restrict__ base,
                                              int ld, int lane) {
    int n  = lane & 15;
    int kk = (lane & 16) ? 16 : 0;
    return *(const v16h*)(base + (size_t)n * ld + kk);
}

// B fragment from the padded LDS panel.
__device__ __forceinline__ v16h load_b_frag_pad(const _Float16* sB, int ntile,
                                                int k0, int lane) {
    int n  = lane & 15;
    int kk = (lane & 16) ? 16 : 0;
    return *(const v16h*)(sB + (ntile * 16 + n) * PADK + k0 + kk);
}

// ---------------------------------------------------------------------------
// f32 -> f16 conversion
// ---------------------------------------------------------------------------
__global__ void cvt_f32_to_f16(const float* __restrict__ src,
                               _Float16* __restrict__ dst, int n) {
    int i = blockIdx.x * blockDim.x + threadIdx.x;
    if (i < n) dst[i] = (_Float16)src[i];
}

// ---------------------------------------------------------------------------
// Tiled WMMA GEMM:  D[M,N] = A[M,K] * W[N,K]^T + bias
// Block: 128 threads = 4 waves; block tile 128(M) x 64(N); wave tile 32x64.
// Whole 64 x K weight panel staged once in LDS -> barrier-free K loop.
// mode 0: scatter-store f16 into q16/k16/v16 ([B,H,N,64] layout)
// mode 1: store f32 + bias into out32 ([M,N] row-major)
// ---------------------------------------------------------------------------
__global__ void gemm_wmma_f16(const _Float16* __restrict__ A,
                              const _Float16* __restrict__ W,
                              const float* __restrict__ bias,
                              int M, int N, int K, int mode,
                              _Float16* __restrict__ q16,
                              _Float16* __restrict__ k16,
                              _Float16* __restrict__ v16,
                              float* __restrict__ out32) {
    extern __shared__ __align__(16) char smem_raw[];
    _Float16* sB = (_Float16*)smem_raw;   // [64][PADK]

    const int tid  = threadIdx.x;
    const int lane = tid & 31;
    const int wave = tid >> 5;
    const int n0blk = blockIdx.x * 64;
    const int row0  = blockIdx.y * 128 + wave * 32;

    // ---- stage full weight panel W[n0blk..+63][0..K) into LDS -------------
    {
        int r   = tid >> 1;               // panel row 0..63
        int sel = tid & 1;                // half of the row
        const _Float16* src = W + (size_t)(n0blk + r) * K + sel * (K / 2);
        _Float16*       dst = sB + r * PADK + sel * (K / 2);
#pragma unroll
        for (int i = 0; i < (K / 2) / 16; ++i)
            *(v16h*)(dst + i * 16) = *(const v16h*)(src + i * 16);
    }
    __syncthreads();

    const _Float16* pA0 = A + (size_t)row0 * K;
    const _Float16* pA1 = A + (size_t)(row0 + 16) * K;

    v8f acc[2][4];
#pragma unroll
    for (int m = 0; m < 2; ++m)
#pragma unroll
        for (int t = 0; t < 4; ++t) acc[m][t] = v8f{};

    // ---- barrier-free K loop ---------------------------------------------
    for (int k0 = 0; k0 < K; k0 += 32) {
        if (k0 + 128 < K) {               // prefetch A four k-steps ahead
            __builtin_prefetch(pA0 + k0 + 128, 0, 1);
            __builtin_prefetch(pA1 + k0 + 128, 0, 1);
        }
        // distinct variables -> distinct VGPRs -> overlapping ds_loads
        v16h b0 = load_b_frag_pad(sB, 0, k0, lane);
        v16h b1 = load_b_frag_pad(sB, 1, k0, lane);
        v16h b2 = load_b_frag_pad(sB, 2, k0, lane);
        v16h b3 = load_b_frag_pad(sB, 3, k0, lane);
        v16h a0 = load_a_frag(pA0 + k0, K, lane);
        v16h a1 = load_a_frag(pA1 + k0, K, lane);

        acc[0][0] = wmma_f16(a0, b0, acc[0][0]);
        acc[1][0] = wmma_f16(a1, b0, acc[1][0]);
        acc[0][1] = wmma_f16(a0, b1, acc[0][1]);
        acc[1][1] = wmma_f16(a1, b1, acc[1][1]);
        acc[0][2] = wmma_f16(a0, b2, acc[0][2]);
        acc[1][2] = wmma_f16(a1, b2, acc[1][2]);
        acc[0][3] = wmma_f16(a0, b3, acc[0][3]);
        acc[1][3] = wmma_f16(a1, b3, acc[1][3]);
    }

    // ---- epilogue: C/D layout -> col = lane&15, row = vgpr + 8*(lane>=16) -
    const int n  = lane & 15;
    const int mb = (lane & 16) ? 8 : 0;
#pragma unroll
    for (int m = 0; m < 2; ++m) {
        const int rbase = row0 + m * 16 + mb;
#pragma unroll
        for (int t = 0; t < 4; ++t) {
            int col = n0blk + t * 16 + n;
            float bv = bias[col];
            if (mode == 0) {
                int which = col / Cc;
                int rem   = col - which * Cc;
                int hh    = rem >> 6;
                int dd    = rem & 63;
                _Float16* dst = (which == 0) ? q16 : (which == 1) ? k16 : v16;
#pragma unroll
                for (int i = 0; i < 8; ++i) {
                    int row = rbase + i;
                    int bb  = row >> 10;
                    int nn  = row & 1023;
                    dst[(((size_t)(bb * Hc + hh) << 10) + nn) * Dc + dd] =
                        (_Float16)(acc[m][t][i] + bv);
                }
            } else {
#pragma unroll
                for (int i = 0; i < 8; ++i) {
                    out32[(size_t)(rbase + i) * N + col] = acc[m][t][i] + bv;
                }
            }
        }
    }
}

// ---------------------------------------------------------------------------
// Block-diagonal attention: one workgroup per (b, h, block).
// 128 threads = 4 waves. Q,K,V tiles are 32x64 f16 in [B,H,N,64] layout.
// ---------------------------------------------------------------------------
__global__ void attn_wmma(const _Float16* __restrict__ q16,
                          const _Float16* __restrict__ k16,
                          const _Float16* __restrict__ v16,
                          _Float16* __restrict__ o16, float scale) {
    const int blk = blockIdx.x;
    const int h   = blockIdx.y;
    const int b   = blockIdx.z;
    const int tid  = threadIdx.x;
    const int lane = tid & 31;
    const int wave = tid >> 5;

    __shared__ __align__(32) float    sS[BSc * BSc];   // logits (f32)
    __shared__ __align__(32) _Float16 sP[BSc * BSc];   // probs  (f16)
    __shared__ __align__(32) _Float16 sVT[Dc * BSc];   // V^T: [d][tok]

    const size_t base = (((size_t)(b * Hc + h)) * Nc + blk * BSc) * Dc;
    const _Float16* Qb = q16 + base;
    const _Float16* Kb = k16 + base;
    const _Float16* Vb = v16 + base;

    // stage V transposed into LDS: sVT[d][tok] = V[tok][d]
    {
        int tok = tid >> 2, seg = tid & 3;
        v16h vv = *(const v16h*)(Vb + tok * Dc + seg * 16);
#pragma unroll
        for (int i = 0; i < 16; ++i) sVT[(seg * 16 + i) * BSc + tok] = vv[i];
    }

    // ---- S = Q K^T : four 16x16 tiles, one per wave -----------------------
    {
        const int mi = wave >> 1, ni = wave & 1;
        v16h a0 = load_a_frag(Qb + (mi * 16) * Dc, Dc, lane);
        v16h a1 = load_a_frag(Qb + (mi * 16) * Dc + 32, Dc, lane);
        v16h b0 = load_b_frag_g(Kb + (ni * 16) * Dc, Dc, lane);
        v16h b1 = load_b_frag_g(Kb + (ni * 16) * Dc + 32, Dc, lane);
        v8f acc = v8f{};
        acc = wmma_f16(a0, b0, acc);
        acc = wmma_f16(a1, b1, acc);
        const int n  = lane & 15;
        const int mb = (lane & 16) ? 8 : 0;
#pragma unroll
        for (int i = 0; i < 8; ++i)
            sS[(mi * 16 + mb + i) * BSc + ni * 16 + n] = acc[i] * scale;
    }
    __syncthreads();

    // ---- row softmax (32 rows, one thread each) ---------------------------
    if (tid < BSc) {
        float m = -1e30f;
        for (int c = 0; c < BSc; ++c) m = fmaxf(m, sS[tid * BSc + c]);
        float s = 0.f;
        for (int c = 0; c < BSc; ++c) {
            float e = __expf(sS[tid * BSc + c] - m);
            sS[tid * BSc + c] = e;
            s += e;
        }
        float r = 1.f / s;
        for (int c = 0; c < BSc; ++c)
            sP[tid * BSc + c] = (_Float16)(sS[tid * BSc + c] * r);
    }
    __syncthreads();

    // ---- O = P V : 2(M) x 4(N) tiles of 16x16, two per wave ---------------
    const int mi = wave >> 1;
#pragma unroll
    for (int t = 0; t < 2; ++t) {
        const int ni = (wave & 1) * 2 + t;
        // A fragment from sP (32-wide K)
        int r  = lane & 15;
        int kk = (lane & 16) ? 16 : 0;
        const _Float16* pa = sP + (mi * 16 + r) * BSc + kk;
        v8h lo = *(const v8h*)pa;
        v8h hi = *(const v8h*)(pa + 8);
        v16h a;
#pragma unroll
        for (int i = 0; i < 8; ++i) { a[i] = lo[i]; a[i + 8] = hi[i]; }
        // B fragment from sVT: B[k][n] = V[k][n] = sVT[n][k]
        int n  = lane & 15;
        int kb = (lane & 16) ? 16 : 0;
        v16h bb = *(const v16h*)(sVT + (ni * 16 + n) * BSc + kb);

        v8f o = v8f{};
        o = wmma_f16(a, bb, o);

        const int mb = (lane & 16) ? 8 : 0;
        _Float16* dst = o16 +
            ((size_t)(b * Nc + blk * BSc + mi * 16 + mb)) * Cc + h * Dc + ni * 16 + n;
#pragma unroll
        for (int i = 0; i < 8; ++i) dst[(size_t)i * Cc] = (_Float16)o[i];
    }
}

// ---------------------------------------------------------------------------
extern "C" void kernel_launch(void* const* d_in, const int* in_sizes, int n_in,
                              void* d_out, int out_size, void* d_ws, size_t ws_size,
                              hipStream_t stream) {
    (void)in_sizes; (void)n_in; (void)out_size; (void)ws_size;

    const float* x      = (const float*)d_in[0];   // [B,N,C]
    const float* qkv_w  = (const float*)d_in[1];   // [3C,C]
    const float* qkv_b  = (const float*)d_in[2];   // [3C]
    const float* proj_w = (const float*)d_in[3];   // [C,C]
    const float* proj_b = (const float*)d_in[4];   // [C]
    float* out = (float*)d_out;                    // [B,N,C]

    char* ws = (char*)d_ws;
    size_t off = 0;
    auto carve = [&](size_t elems) {
        _Float16* p = (_Float16*)(ws + off);
        off += ((elems * sizeof(_Float16)) + 255) & ~(size_t)255;
        return p;
    };
    _Float16* x16  = carve((size_t)MROWS * Cc);   // 48 MB
    _Float16* w16  = carve((size_t)C3 * Cc);      // 3.4 MB
    _Float16* pw16 = carve((size_t)Cc * Cc);      // 1.1 MB
    _Float16* q16  = carve((size_t)MROWS * Cc);   // 48 MB  [B,H,N,64]
    _Float16* k16  = carve((size_t)MROWS * Cc);   // 48 MB
    _Float16* v16  = carve((size_t)MROWS * Cc);   // 48 MB
    _Float16* o16  = carve((size_t)MROWS * Cc);   // 48 MB  [B,N,C]

    // 1) precision staging f32 -> f16
    {
        int n1 = MROWS * Cc;
        cvt_f32_to_f16<<<(n1 + 255) / 256, 256, 0, stream>>>(x, x16, n1);
        int n2 = C3 * Cc;
        cvt_f32_to_f16<<<(n2 + 255) / 256, 256, 0, stream>>>(qkv_w, w16, n2);
        int n3 = Cc * Cc;
        cvt_f32_to_f16<<<(n3 + 255) / 256, 256, 0, stream>>>(proj_w, pw16, n3);
    }

    // 2) QKV GEMM: [32768 x 768] x [768 x 2304]^T, scatter to q/k/v
    gemm_wmma_f16<<<dim3(C3 / 64, MROWS / 128), 128, SB_BYTES, stream>>>(
        x16, w16, qkv_b, MROWS, C3, Cc, /*mode=*/0, q16, k16, v16, nullptr);

    // 3) block-diagonal attention, one WG per (b,h,block)
    attn_wmma<<<dim3(Nc / BSc, Hc, Bc), 128, 0, stream>>>(
        q16, k16, v16, o16, 0.125f /* 1/sqrt(64) */);

    // 4) output projection: [32768 x 768] x [768 x 768]^T + bias -> f32 out
    gemm_wmma_f16<<<dim3(Cc / 64, MROWS / 128), 128, SB_BYTES, stream>>>(
        o16, pw16, proj_b, MROWS, Cc, Cc, /*mode=*/1,
        nullptr, nullptr, nullptr, out);
}
